// RWKV_18648747999323
// MI455X (gfx1250) — matrix operand order repaired and checked by reading, not compiled, not run on hardware
//
#include <hip/hip_runtime.h>
#include <hip/hip_bf16.h>

// ---------------- problem constants (match reference) ----------------
static constexpr int LNUM = 4;
static constexpr int BB   = 4;
static constexpr int TT   = 2048;
static constexpr int CC   = 1024;
static constexpr int FF   = 4 * CC;           // 4096
static constexpr int MROWS = BB * TT;         // 8192
static constexpr float EPSLN = 1e-5f;
static constexpr float NEG_INF = -1e38f;

// ---------------- vector types for WMMA ----------------
typedef __attribute__((ext_vector_type(16))) __bf16          v16bf;
typedef __attribute__((ext_vector_type(8)))  float           v8f;
typedef __attribute__((ext_vector_type(8)))  unsigned short  ushort8;

// ---------------- helpers ----------------
__device__ __forceinline__ unsigned short f2bf(float f) {
    unsigned int u = __float_as_uint(f);
    unsigned int r = u + 0x7FFFu + ((u >> 16) & 1u);   // round-to-nearest-even
    return (unsigned short)(r >> 16);
}
__device__ __forceinline__ float sigmoidf_fast(float x) {
    return 1.0f / (1.0f + __expf(-x));
}

// CDNA5 async global->LDS copy (ASYNCcnt-tracked, no VGPR round-trip).
// LDS operand is the DS-relative byte address == low 32 bits of the flat
// pointer to a __shared__ object (gfx1250 aperture mapping).
__device__ __forceinline__ void async_copy_b128(unsigned short* lds_dst,
                                                const unsigned short* gsrc) {
    const unsigned int ldsoff = (unsigned int)(unsigned long long)lds_dst;
    asm volatile("global_load_async_to_lds_b128 %0, %1, off"
                 :: "v"(ldsoff), "v"(gsrc)
                 : "memory");
}

// ===================================================================
// LayerNorm: one block per row, C=1024, 256 threads (8 wave32s)
// ===================================================================
__global__ __launch_bounds__(256)
void rwkv_ln_kernel(const float* __restrict__ x, const float* __restrict__ w,
                    float* __restrict__ y) {
    const int row = blockIdx.x;
    const float* xr = x + (size_t)row * CC;
    float* yr = y + (size_t)row * CC;
    __shared__ float red[8];
    __shared__ float sstat[2];

    // mean
    float s = 0.f;
    for (int c = threadIdx.x; c < CC; c += blockDim.x) s += xr[c];
#pragma unroll
    for (int o = 16; o > 0; o >>= 1) s += __shfl_down(s, o, 32);
    if ((threadIdx.x & 31) == 0) red[threadIdx.x >> 5] = s;
    __syncthreads();
    if (threadIdx.x == 0) {
        float t = 0.f;
#pragma unroll
        for (int i = 0; i < 8; ++i) t += red[i];
        sstat[0] = t / (float)CC;
    }
    __syncthreads();
    const float mu = sstat[0];

    // variance
    float s2 = 0.f;
    for (int c = threadIdx.x; c < CC; c += blockDim.x) {
        float d = xr[c] - mu;
        s2 += d * d;
    }
#pragma unroll
    for (int o = 16; o > 0; o >>= 1) s2 += __shfl_down(s2, o, 32);
    if ((threadIdx.x & 31) == 0) red[threadIdx.x >> 5] = s2;
    __syncthreads();
    if (threadIdx.x == 0) {
        float t = 0.f;
#pragma unroll
        for (int i = 0; i < 8; ++i) t += red[i];
        sstat[1] = rsqrtf(t / (float)CC + EPSLN);
    }
    __syncthreads();
    const float rstd = sstat[1];

    for (int c = threadIdx.x; c < CC; c += blockDim.x)
        yr[c] = (xr[c] - mu) * rstd * w[c];
}

// ===================================================================
// Time-mix: xk/xv/xr = h*m + timeshift(h)*(1-m), output bf16
// ===================================================================
__global__ __launch_bounds__(256)
void rwkv_tmmix_kernel(const float* __restrict__ h,
                       const float* __restrict__ mk, const float* __restrict__ mv,
                       const float* __restrict__ mr,
                       unsigned short* __restrict__ ak, unsigned short* __restrict__ av,
                       unsigned short* __restrict__ ar) {
    const long i = (long)blockIdx.x * blockDim.x + threadIdx.x;   // < B*T*C
    const int c = (int)(i & (CC - 1));
    const int t = (int)((i / CC) % TT);
    const float hv = h[i];
    const float pv = (t == 0) ? 0.f : h[i - CC];
    float m;
    m = mk[c]; ak[i] = f2bf(hv * m + pv * (1.f - m));
    m = mv[c]; av[i] = f2bf(hv * m + pv * (1.f - m));
    m = mr[c]; ar[i] = f2bf(hv * m + pv * (1.f - m));
}

// channel-mix variant: only k and r
__global__ __launch_bounds__(256)
void rwkv_cmmix_kernel(const float* __restrict__ h,
                       const float* __restrict__ mk, const float* __restrict__ mr,
                       unsigned short* __restrict__ ak, unsigned short* __restrict__ ar) {
    const long i = (long)blockIdx.x * blockDim.x + threadIdx.x;
    const int c = (int)(i & (CC - 1));
    const int t = (int)((i / CC) % TT);
    const float hv = h[i];
    const float pv = (t == 0) ? 0.f : h[i - CC];
    float m;
    m = mk[c]; ak[i] = f2bf(hv * m + pv * (1.f - m));
    m = mr[c]; ar[i] = f2bf(hv * m + pv * (1.f - m));
}

// ===================================================================
// WKV scan: one thread per (b,c) channel; sequential over T. In-place
// v -> wkv (each thread owns its column).
// ===================================================================
__global__ __launch_bounds__(256)
void rwkv_wkv_kernel(const float* __restrict__ k, float* __restrict__ v,
                     const float* __restrict__ td, const float* __restrict__ tf) {
    const int idx = blockIdx.x * blockDim.x + threadIdx.x;   // < B*C
    if (idx >= BB * CC) return;
    const int b = idx / CC, c = idx % CC;
    const float w = -__expf(td[c]);
    const float u = tf[c];
    float num = 0.f, den = 0.f, mx = NEG_INF;
    const size_t base = (size_t)b * TT * CC + c;
    for (int t = 0; t < TT; ++t) {
        const size_t o = base + (size_t)t * CC;
        const float kt = k[o];
        const float vt = v[o];
        float mo = fmaxf(mx, kt + u);
        float e1 = __expf(mx - mo), e2 = __expf(kt + u - mo);
        const float out = (e1 * num + e2 * vt) / (e1 * den + e2);
        const float ms = fmaxf(mx + w, kt);
        e1 = __expf(mx + w - ms);
        e2 = __expf(kt - ms);
        num = e1 * num + e2 * vt;
        den = e1 * den + e2;
        mx = ms;
        v[o] = out;
    }
}

// elementwise r*wkv -> bf16 (GEMM A input)
__global__ __launch_bounds__(256)
void rwkv_rmul_kernel(const float* __restrict__ r, const float* __restrict__ wkv,
                      unsigned short* __restrict__ a) {
    const long i = (long)blockIdx.x * blockDim.x + threadIdx.x;
    a[i] = f2bf(r[i] * wkv[i]);
}

// fp32 weights -> bf16 staging
__global__ __launch_bounds__(256)
void rwkv_cvt_kernel(const float* __restrict__ in, unsigned short* __restrict__ out, int n) {
    const int i = blockIdx.x * blockDim.x + threadIdx.x;
    if (i < n) out[i] = f2bf(in[i]);
}

// ===================================================================
// bf16 WMMA GEMM:  Y[M,N] = A[M,K] * W[N,K]^T   (einsum 'mk,nk->mn')
// 128x128 block tile, KB=32, 256 threads = 8 wave32s (2 M-waves x 4 N-waves),
// each wave computes 4x2 = 8 accumulators of 16x16 via v_wmma_f32_16x16x32_bf16.
// Global->LDS staging uses CDNA5 async-to-LDS copies with 2-deep LDS double
// buffering: tile i+1's HBM/L2 latency overlaps tile i's WMMA compute.
// Epilogues: 0 store f32 | 1 sigmoid f32 | 2 relu^2 -> bf16 | 3 out+=y | 4 out+=scale*y
// ===================================================================
static constexpr int GBM = 128, GBN = 128, GKB = 32;
static constexpr int LDS_S = GKB + 8;   // 40 bf16 per row (16B-aligned rows)

template <int EPI>
__global__ __launch_bounds__(256)
void rwkv_gemm_kernel(const unsigned short* __restrict__ A,
                      const unsigned short* __restrict__ W,
                      float* __restrict__ outf, unsigned short* __restrict__ outb,
                      const float* __restrict__ scale,
                      int M, int N, int K) {
    __shared__ __align__(16) unsigned short As[2][GBM * LDS_S];
    __shared__ __align__(16) unsigned short Bs[2][GBN * LDS_S];

    const int tid  = threadIdx.x;
    const int lane = tid & 31;
    const int wv   = tid >> 5;
    const int wm   = wv & 1;        // 0..1 : 64 M-rows per wave-row
    const int wn   = wv >> 1;       // 0..3 : 32 N-cols per wave-col
    const int l15  = lane & 15;
    const int hf   = lane >> 4;

    const int m0 = blockIdx.y * GBM;
    const int n0 = blockIdx.x * GBN;

    const int lrow   = tid >> 2;         // 0..63
    const int lchunk = (tid & 3) * 8;    // 0,8,16,24 (bf16 elems)

    v8f acc[4][2] = {};

    // issue one tile's async copies (4 x b128 per thread -> +4 ASYNCcnt per wave)
    auto issue_tile = [&](int kt, int buf) {
#pragma unroll
        for (int p = 0; p < 2; ++p) {
            const int row = lrow + p * 64;
            async_copy_b128(&As[buf][row * LDS_S + lchunk],
                            &A[(size_t)(m0 + row) * K + kt * GKB + lchunk]);
            async_copy_b128(&Bs[buf][row * LDS_S + lchunk],
                            &W[(size_t)(n0 + row) * K + kt * GKB + lchunk]);
        }
    };

    const int ntiles = K / GKB;
    issue_tile(0, 0);

    for (int kt = 0; kt < ntiles; ++kt) {
        const int buf = kt & 1;

        if (kt + 1 < ntiles) {
            issue_tile(kt + 1, buf ^ 1);
            // prefetch tile kt+2 toward L2 (global_prefetch_b8)
            if (kt + 2 < ntiles) {
                __builtin_prefetch(&A[(size_t)(m0 + lrow) * K + (kt + 2) * GKB + lchunk]);
                __builtin_prefetch(&W[(size_t)(n0 + lrow) * K + (kt + 2) * GKB + lchunk]);
            }
            // ASYNCcnt decrements in order: <=4 means tile kt's 4 copies landed,
            // tile kt+1's 4 copies stay in flight.
            asm volatile("s_wait_asynccnt 0x4" ::: "memory");
        } else {
            asm volatile("s_wait_asynccnt 0x0" ::: "memory");
        }
        __syncthreads();   // all waves' portions of tile kt visible in LDS

        // ---- assemble fragments per ISA 16-bit layouts ----
        v16bf afr[4], bfr[2];
#pragma unroll
        for (int i = 0; i < 4; ++i) {
            const int r = wm * 64 + i * 16 + l15;
            // A 16x32 bf16: lanes 0-15 K={0..7,16..23}, lanes 16-31 K={8..15,24..31}
            ushort8 lo = *reinterpret_cast<const ushort8*>(&As[buf][r * LDS_S + 8 * hf]);
            ushort8 hi = *reinterpret_cast<const ushort8*>(&As[buf][r * LDS_S + 16 + 8 * hf]);
            afr[i] = __builtin_bit_cast(v16bf,
                __builtin_shufflevector(lo, hi, 0,1,2,3,4,5,6,7,8,9,10,11,12,13,14,15));
        }
#pragma unroll
        for (int j = 0; j < 2; ++j) {
            const int r = wn * 32 + j * 16 + l15;
            // B 32x16 bf16: lanes 0-15 K={0..7,16..23}, lanes 16-31 K={8..15,24..31}
            ushort8 lo = *reinterpret_cast<const ushort8*>(&Bs[buf][r * LDS_S + 8 * hf]);
            ushort8 hi = *reinterpret_cast<const ushort8*>(&Bs[buf][r * LDS_S + 16 + 8 * hf]);
            bfr[j] = __builtin_bit_cast(v16bf,
                __builtin_shufflevector(lo, hi, 0,1,2,3,4,5,6,7,8,9,10,11,12,13,14,15));
        }

#pragma unroll
        for (int i = 0; i < 4; ++i)
#pragma unroll
            for (int j = 0; j < 2; ++j)
                acc[i][j] = __builtin_amdgcn_wmma_f32_16x16x32_bf16(
                    false, afr[i], false, bfr[j], (short)0, acc[i][j], false, false);

        __syncthreads();   // everyone done reading buf before it is re-filled
    }

    // ---- epilogue: C/D layout: lane l -> n = l&15; VGPR r -> m = r + 8*(l>>4) ----
#pragma unroll
    for (int i = 0; i < 4; ++i) {
#pragma unroll
        for (int j = 0; j < 2; ++j) {
            const int gmb = m0 + wm * 64 + i * 16 + hf * 8;
            const int gn  = n0 + wn * 32 + j * 16 + l15;
            const v8f d = acc[i][j];
#pragma unroll
            for (int r = 0; r < 8; ++r) {
                const float v = d[r];
                const size_t o = (size_t)(gmb + r) * N + gn;
                if constexpr (EPI == 0) {
                    outf[o] = v;
                } else if constexpr (EPI == 1) {
                    outf[o] = sigmoidf_fast(v);
                } else if constexpr (EPI == 2) {
                    const float t = v > 0.f ? v : 0.f;
                    outb[o] = f2bf(t * t);
                } else if constexpr (EPI == 3) {
                    outf[o] += v;
                } else {  // EPI == 4
                    outf[o] += scale[o] * v;
                }
            }
        }
    }
}

// ---------------- host-side launch helpers ----------------
template <int EPI>
static void gemm_launch(const unsigned short* A, const unsigned short* W,
                        float* outf, unsigned short* outb, const float* scale,
                        int M, int N, int K, hipStream_t s) {
    dim3 grid(N / GBN, M / GBM);
    rwkv_gemm_kernel<EPI><<<grid, 256, 0, s>>>(A, W, outf, outb, scale, M, N, K);
}

extern "C" void kernel_launch(void* const* d_in, const int* in_sizes, int n_in,
                              void* d_out, int out_size, void* d_ws, size_t ws_size,
                              hipStream_t stream) {
    (void)in_sizes; (void)n_in; (void)out_size; (void)ws_size;

    // inputs, in setup_inputs() dict order (all fp32)
    const float* x_in      = (const float*)d_in[0];
    const float* ln1_w     = (const float*)d_in[1];
    const float* tm_mix_k  = (const float*)d_in[2];
    const float* tm_mix_v  = (const float*)d_in[3];
    const float* tm_mix_r  = (const float*)d_in[4];
    const float* tm_k_w    = (const float*)d_in[5];
    const float* tm_v_w    = (const float*)d_in[6];
    const float* tm_r_w    = (const float*)d_in[7];
    const float* tm_o_w    = (const float*)d_in[8];
    const float* time_dec  = (const float*)d_in[9];
    const float* time_fst  = (const float*)d_in[10];
    const float* ln2_w     = (const float*)d_in[11];
    const float* cm_mix_k  = (const float*)d_in[12];
    const float* cm_mix_r  = (const float*)d_in[13];
    const float* cm_k_w    = (const float*)d_in[14];
    const float* cm_v_w    = (const float*)d_in[15];
    const float* cm_r_w    = (const float*)d_in[16];
    const float* lnf_w     = (const float*)d_in[17];

    // ---------------- workspace layout ----------------
    const size_t BTC = (size_t)BB * TT * CC;   // 8,388,608
    const size_t BTF = (size_t)BB * TT * FF;   // 33,554,432
    float* xbuf = (float*)d_ws;                // residual stream
    float* hbuf = xbuf + BTC;                  // LN output
    float* kbuf = hbuf + BTC;                  // k
    float* vbuf = kbuf + BTC;                  // v, then wkv in-place
    float* rbuf = vbuf + BTC;                  // r / rr (post-sigmoid)
    unsigned short* abuf  = (unsigned short*)(rbuf + BTC); // bf16 GEMM-A arena (BTF elems)
    unsigned short* kkbuf = abuf + BTF;                    // bf16 relu(kk)^2 (BTF elems)
    unsigned short* wbuf  = kkbuf + BTF;                   // bf16 weight staging (FF*CC elems)

    unsigned short* ak = abuf;
    unsigned short* av = abuf + BTC;
    unsigned short* ar = abuf + 2 * BTC;

    const int EW_BLK = 256;
    const int grid_btc = (int)(BTC / EW_BLK);
    const int grid_cc  = (CC * CC) / EW_BLK;
    const int grid_fc  = (FF * CC) / EW_BLK;

    // x (residual) := input
    hipMemcpyAsync(xbuf, x_in, BTC * sizeof(float), hipMemcpyDeviceToDevice, stream);

    for (int l = 0; l < LNUM; ++l) {
        // ---------------- TimeMixing ----------------
        rwkv_ln_kernel<<<MROWS, 256, 0, stream>>>(xbuf, ln1_w + l * CC, hbuf);
        rwkv_tmmix_kernel<<<grid_btc, EW_BLK, 0, stream>>>(
            hbuf, tm_mix_k + l * CC, tm_mix_v + l * CC, tm_mix_r + l * CC, ak, av, ar);

        rwkv_cvt_kernel<<<grid_cc, EW_BLK, 0, stream>>>(tm_k_w + (size_t)l * CC * CC, wbuf, CC * CC);
        gemm_launch<0>(ak, wbuf, kbuf, nullptr, nullptr, MROWS, CC, CC, stream);

        rwkv_cvt_kernel<<<grid_cc, EW_BLK, 0, stream>>>(tm_v_w + (size_t)l * CC * CC, wbuf, CC * CC);
        gemm_launch<0>(av, wbuf, vbuf, nullptr, nullptr, MROWS, CC, CC, stream);

        rwkv_cvt_kernel<<<grid_cc, EW_BLK, 0, stream>>>(tm_r_w + (size_t)l * CC * CC, wbuf, CC * CC);
        gemm_launch<1>(ar, wbuf, rbuf, nullptr, nullptr, MROWS, CC, CC, stream);

        rwkv_wkv_kernel<<<(BB * CC) / 256, 256, 0, stream>>>(
            kbuf, vbuf, time_dec + l * CC, time_fst + l * CC);

        rwkv_rmul_kernel<<<grid_btc, EW_BLK, 0, stream>>>(rbuf, vbuf, abuf);
        rwkv_cvt_kernel<<<grid_cc, EW_BLK, 0, stream>>>(tm_o_w + (size_t)l * CC * CC, wbuf, CC * CC);
        gemm_launch<3>(abuf, wbuf, xbuf, nullptr, nullptr, MROWS, CC, CC, stream);  // x += y

        // ---------------- ChannelMixing ----------------
        rwkv_ln_kernel<<<MROWS, 256, 0, stream>>>(xbuf, ln2_w + l * CC, hbuf);
        rwkv_cmmix_kernel<<<grid_btc, EW_BLK, 0, stream>>>(
            hbuf, cm_mix_k + l * CC, cm_mix_r + l * CC, ak, ar);

        rwkv_cvt_kernel<<<grid_fc, EW_BLK, 0, stream>>>(cm_k_w + (size_t)l * FF * CC, wbuf, FF * CC);
        gemm_launch<2>(ak, wbuf, nullptr, kkbuf, nullptr, MROWS, FF, CC, stream);   // relu^2 -> bf16

        rwkv_cvt_kernel<<<grid_cc, EW_BLK, 0, stream>>>(cm_r_w + (size_t)l * CC * CC, wbuf, CC * CC);
        gemm_launch<1>(ar, wbuf, rbuf, nullptr, nullptr, MROWS, CC, CC, stream);    // sigmoid

        rwkv_cvt_kernel<<<grid_fc, EW_BLK, 0, stream>>>(cm_v_w + (size_t)l * CC * FF, wbuf, CC * FF);
        gemm_launch<4>(kkbuf, wbuf, xbuf, nullptr, rbuf, MROWS, CC, FF, stream);    // x += rr*vv
    }

    // final LayerNorm -> d_out (fp32)
    rwkv_ln_kernel<<<MROWS, 256, 0, stream>>>(xbuf, lnf_w, (float*)d_out);
}
